// GCNBackbone_44452911513915
// MI455X (gfx1250) — matrix operand order repaired and checked by reading, not compiled, not run on hardware
//
#include <hip/hip_runtime.h>
#include <hip/hip_bf16.h>

// ---------------------------------------------------------------------------
// GNN backbone for gfx1250 (MI455X), wave32 + WMMA bf16 (fp32 accumulate).
// Message passing restructured via linearity:
//   segment_sum(concat(h[src],eattr) @ W + b) == [agg_h | eagg] @ W + deg*b
// so edge-dim aggregate (eagg) and degree are computed once; per layer we only
// scatter-add h[src] (fp32 atomics into L2-resident 64MB buffer) + 1 GEMM.
// GEMM: one wave per 16x64 output strip (4 WMMA column tiles); per K-step all
// 10 b128 loads (A pair + 4 B pairs) are issued before any WMMA so the
// scheduler can use partial load waits instead of full drains.
// ---------------------------------------------------------------------------

typedef __attribute__((ext_vector_type(16))) __bf16 v16bf;
typedef __attribute__((ext_vector_type(8)))  __bf16 v8bf;
typedef __attribute__((ext_vector_type(8)))  float  v8f;

#define NN   65536
#define EE   262144
#define HID  256

enum { EPI_NONE = 0, EPI_RELU = 1, EPI_TANH = 2, EPI_DEG = 3 };

// ---------------------------------------------------------------------------
// WMMA GEMM:  C[M,Ncol](fp32, row stride ldc) = A[M,K](bf16) * Wt + epilogue
// Wt is pre-transposed to [Ncol,K] bf16 so B lane loads are 2x contiguous 16B.
// NT column tiles per wave; K-step 32.
// ---------------------------------------------------------------------------
template<int NT>
__global__ __launch_bounds__(256)
void wmma_gemm_k(const __bf16* __restrict__ A, const __bf16* __restrict__ Wt,
                 const float* __restrict__ bias, const float* __restrict__ deg,
                 float* __restrict__ C, int M, int K, int Ncol, int ldc, int mode)
{
    const int wave = blockIdx.x * (blockDim.x >> 5) + (threadIdx.x >> 5);
    const int lane = threadIdx.x & 31;
    const int nTiles  = Ncol >> 4;
    const int nGroups = (nTiles + NT - 1) / NT;
    const int totalGroups = (M >> 4) * nGroups;
    if (wave >= totalGroups) return;
    const int mTile  = wave / nGroups;
    const int nTile0 = (wave - mTile * nGroups) * NT;

    const int hi   = lane >> 4;            // half-wave select
    const int mRow = (mTile << 4) + (lane & 15);
    const int akb  = hi << 4;              // A: K base 0 or 16 (matches B layout)
    const int bkb  = hi << 4;              // B: K base 0 or 16 (contiguous k = 16h + s, same as A)

    const __bf16* aRow = A + (size_t)mRow * K;
    const __bf16* bRow[NT];
    #pragma unroll
    for (int j = 0; j < NT; ++j) {
        int nt = nTile0 + j;
        if (nt >= nTiles) nt = nTiles - 1;           // clamp: in-bounds dummy tile
        bRow[j] = Wt + (size_t)((nt << 4) + (lane & 15)) * K;
    }

    v8f acc[NT];
    #pragma unroll
    for (int j = 0; j < NT; ++j) acc[j] = (v8f){};

    for (int k0 = 0; k0 < K; k0 += 32) {
        // Issue ALL loads for this K-step first (distinct live ranges) so the
        // backend can clause them and partially overlap with the WMMAs below.
        v8bf a0 = *(const v8bf*)(aRow + k0 + akb);        // K = akb .. akb+7
        v8bf a1 = *(const v8bf*)(aRow + k0 + akb + 8);    // K = akb+8 .. akb+15
        v8bf b0[NT], b1[NT];
        #pragma unroll
        for (int j = 0; j < NT; ++j) {
            b0[j] = *(const v8bf*)(bRow[j] + k0 + bkb);      // K = bkb .. bkb+7
            b1[j] = *(const v8bf*)(bRow[j] + k0 + bkb + 8);  // K = bkb+8 .. bkb+15
        }
        v16bf av;
        #pragma unroll
        for (int i = 0; i < 8; ++i) { av[i] = a0[i]; av[i + 8] = a1[i]; }
        #pragma unroll
        for (int j = 0; j < NT; ++j) {
            v16bf bv;
            #pragma unroll
            for (int i = 0; i < 8; ++i) { bv[i] = b0[j][i]; bv[i + 8] = b1[j][i]; }
            acc[j] = __builtin_amdgcn_wmma_f32_16x16x32_bf16(
                         false, av, false, bv, (short)0, acc[j], false, false);
        }
    }

    #pragma unroll
    for (int j = 0; j < NT; ++j) {
        const int nt = nTile0 + j;
        if (nt < nTiles) {                              // wave-uniform guard
            const int nCol = (nt << 4) + (lane & 15);
            const float bval = bias ? bias[nCol] : 0.0f;
            #pragma unroll
            for (int r = 0; r < 8; ++r) {
                const int row = (mTile << 4) + r + (hi << 3);   // ISA C/D layout
                float v = acc[j][r];
                if (mode == EPI_DEG)       v += deg[row] * bval;
                else                       v += bval;
                if (mode == EPI_RELU)      v = fmaxf(v, 0.0f);
                else if (mode == EPI_TANH) v = tanhf(v);
                C[(size_t)row * ldc + nCol] = v;
            }
        }
    }
}

// Transpose + fp32->bf16:  out[n*K+k] = in[k*Ncol+n]   (weights)
__global__ __launch_bounds__(256)
void transcvt_k(const float* __restrict__ in, __bf16* __restrict__ out, int K, int Ncol)
{
    int idx = blockIdx.x * 256 + threadIdx.x;
    if (idx >= K * Ncol) return;
    int n = idx / K, k = idx - n * K;
    out[idx] = (__bf16)in[(size_t)k * Ncol + n];
}

// Flat fp32 -> bf16
__global__ __launch_bounds__(256)
void f2bf_k(const float* __restrict__ in, __bf16* __restrict__ out, int total)
{
    int idx = blockIdx.x * 256 + threadIdx.x;
    if (idx < total) out[idx] = (__bf16)in[idx];
}

// Encoder LayerNorm: in fp32 [M,256] -> bf16 out (one wave per row)
__global__ __launch_bounds__(256)
void ln_enc_k(const float* __restrict__ in, const float* __restrict__ g,
              const float* __restrict__ b, __bf16* __restrict__ out, int M)
{
    int row = blockIdx.x * 8 + (threadIdx.x >> 5);
    int lane = threadIdx.x & 31;
    if (row >= M) return;
    const float* r = in + (size_t)row * HID;
    float v[8], s = 0.f;
    #pragma unroll
    for (int i = 0; i < 8; ++i) { v[i] = r[lane + 32 * i]; s += v[i]; }
    #pragma unroll
    for (int o = 16; o; o >>= 1) s += __shfl_xor(s, o, 32);
    float mean = s * (1.0f / HID), var = 0.f;
    #pragma unroll
    for (int i = 0; i < 8; ++i) { float d = v[i] - mean; var += d * d; }
    #pragma unroll
    for (int o = 16; o; o >>= 1) var += __shfl_xor(var, o, 32);
    float rs = rsqrtf(var * (1.0f / HID) + 1e-5f);
    #pragma unroll
    for (int i = 0; i < 8; ++i) {
        int c = lane + 32 * i;
        out[(size_t)row * HID + c] = (__bf16)((v[i] - mean) * rs * g[c] + b[c]);
    }
}

// Conv epilogue: LayerNorm -> SiLU -> (residual) ; writes h fp32 + hb bf16
__global__ __launch_bounds__(256)
void ln_silu_res_k(const float* __restrict__ pre, const float* __restrict__ g,
                   const float* __restrict__ b, float* __restrict__ h,
                   __bf16* __restrict__ hb, int M, int first)
{
    int row = blockIdx.x * 8 + (threadIdx.x >> 5);
    int lane = threadIdx.x & 31;
    if (row >= M) return;
    const float* r = pre + (size_t)row * HID;
    float v[8], s = 0.f;
    #pragma unroll
    for (int i = 0; i < 8; ++i) { v[i] = r[lane + 32 * i]; s += v[i]; }
    #pragma unroll
    for (int o = 16; o; o >>= 1) s += __shfl_xor(s, o, 32);
    float mean = s * (1.0f / HID), var = 0.f;
    #pragma unroll
    for (int i = 0; i < 8; ++i) { float d = v[i] - mean; var += d * d; }
    #pragma unroll
    for (int o = 16; o; o >>= 1) var += __shfl_xor(var, o, 32);
    float rs = rsqrtf(var * (1.0f / HID) + 1e-5f);
    #pragma unroll
    for (int i = 0; i < 8; ++i) {
        int c = lane + 32 * i;
        float y = (v[i] - mean) * rs * g[c] + b[c];
        float sil = y / (1.0f + expf(-y));
        size_t idx = (size_t)row * HID + c;
        float nv = first ? sil : h[idx] + sil;
        h[idx] = nv;
        hb[idx] = (__bf16)nv;
    }
}

// h[:,224:256] = time_emb[batch_idx]
__global__ __launch_bounds__(256)
void timecat_k(const float* __restrict__ time_emb, float* __restrict__ h)
{
    int idx = blockIdx.x * 256 + threadIdx.x;      // N*32
    int n = idx >> 5, c = idx & 31;
    h[(size_t)n * HID + 224 + c] = time_emb[(n >> 11) * 32 + c];   // batch = n/2048
}

// Once: in-degree + segment_sum(edge_attr, dst).  One wave per edge (32 cols).
__global__ __launch_bounds__(256)
void edge_deg_k(const int* __restrict__ ei, const float* __restrict__ eattr,
                float* __restrict__ eagg, float* __restrict__ deg, int E)
{
    int e = blockIdx.x * 8 + (threadIdx.x >> 5);
    int lane = threadIdx.x & 31;
    if (e >= E) return;
    int dst = ei[E + e];
    unsafeAtomicAdd(&eagg[(size_t)dst * 32 + lane], eattr[(size_t)e * 32 + lane]);
    if (lane == 0) unsafeAtomicAdd(&deg[dst], 1.0f);
}

// Once: aggb[:,256:288] = bf16(eagg)
__global__ __launch_bounds__(256)
void eagg_cat_k(const float* __restrict__ eagg, __bf16* __restrict__ aggb)
{
    int idx = blockIdx.x * 256 + threadIdx.x;      // N*32
    int n = idx >> 5, c = idx & 31;
    aggb[(size_t)n * 288 + 256 + c] = (__bf16)eagg[idx];
}

// Per layer: agg[dst] += h[src]  (256 cols, one wave per edge)
__global__ __launch_bounds__(256)
void scatter_k(const int* __restrict__ ei, const float* __restrict__ h,
               float* __restrict__ agg, int E)
{
    int e = blockIdx.x * 8 + (threadIdx.x >> 5);
    int lane = threadIdx.x & 31;
    if (e >= E) return;
    int src = ei[e], dst = ei[E + e];
    const float* hs = h + (size_t)src * HID;
    float* ad = agg + (size_t)dst * HID;
    #pragma unroll
    for (int i = 0; i < 8; ++i)
        unsafeAtomicAdd(&ad[lane + 32 * i], hs[lane + 32 * i]);
}

// Per layer: aggb[:,0:256] = bf16(agg)
__global__ __launch_bounds__(256)
void aggcvt_k(const float* __restrict__ agg, __bf16* __restrict__ aggb)
{
    int idx = blockIdx.x * 256 + threadIdx.x;      // N*256
    int n = idx >> 8, c = idx & 255;
    aggb[(size_t)n * 288 + c] = (__bf16)agg[idx];
}

// score[n] = dot(s1[n,:128], W_a2) + b_a2   (one wave per node)
__global__ __launch_bounds__(256)
void score_k(const float* __restrict__ s1, const float* __restrict__ wa2,
             const float* __restrict__ ba2, float* __restrict__ score, int M)
{
    int row = blockIdx.x * 8 + (threadIdx.x >> 5);
    int lane = threadIdx.x & 31;
    if (row >= M) return;
    const float* r = s1 + (size_t)row * 128;
    float s = 0.f;
    #pragma unroll
    for (int i = 0; i < 4; ++i) s += r[lane + 32 * i] * wa2[lane + 32 * i];
    #pragma unroll
    for (int o = 16; o; o >>= 1) s += __shfl_xor(s, o, 32);
    if (lane == 0) score[row] = s + ba2[0];
}

// Attention pool: each group = (b,t,type) has exactly 32 nodes (n = base+4j).
// Block per group: wave 0 does the segment softmax, 256 threads do weighted sum.
__global__ __launch_bounds__(256)
void pool_k(const float* __restrict__ h, const float* __restrict__ score,
            float* __restrict__ pooled)
{
    __shared__ float attn[32];
    int g = blockIdx.x;
    int b = g >> 6, rem = g & 63, t = rem >> 2, type = rem & 3;
    int base = b * 2048 + t * 128 + type;
    int tid = threadIdx.x;
    if (tid < 32) {
        float s = score[base + 4 * tid];
        float m = s;
        #pragma unroll
        for (int o = 16; o; o >>= 1) m = fmaxf(m, __shfl_xor(m, o, 32));
        float e = expf(s - m), den = e;
        #pragma unroll
        for (int o = 16; o; o >>= 1) den += __shfl_xor(den, o, 32);
        attn[tid] = e / den;
    }
    __syncthreads();
    float acc = 0.f;
    #pragma unroll 4
    for (int j = 0; j < 32; ++j)
        acc += h[(size_t)(base + 4 * j) * HID + tid] * attn[j];
    pooled[(size_t)g * HID + tid] = acc;
}

// Gated fusion over 4 types per (b,t): block per (b,t)
__global__ __launch_bounds__(256)
void fuse_k(const float* __restrict__ pooled, const float* __restrict__ wg,
            const float* __restrict__ bg, float* __restrict__ out)
{
    __shared__ float logits[4];
    int bt = blockIdx.x, tid = threadIdx.x;
    int wv = tid >> 5, lane = tid & 31;
    if (wv < 4) {
        const float* p = pooled + (size_t)(bt * 4 + wv) * HID;
        float s = 0.f;
        #pragma unroll
        for (int i = 0; i < 8; ++i) s += p[lane + 32 * i] * wg[lane + 32 * i];
        #pragma unroll
        for (int o = 16; o; o >>= 1) s += __shfl_xor(s, o, 32);
        if (lane == 0) logits[wv] = s + bg[0];
    }
    __syncthreads();
    float l0 = logits[0], l1 = logits[1], l2 = logits[2], l3 = logits[3];
    float m = fmaxf(fmaxf(l0, l1), fmaxf(l2, l3));
    float e0 = expf(l0 - m), e1 = expf(l1 - m), e2 = expf(l2 - m), e3 = expf(l3 - m);
    float inv = 1.0f / (e0 + e1 + e2 + e3);
    const float* p = pooled + (size_t)bt * 4 * HID;
    out[(size_t)bt * HID + tid] =
        (p[tid] * e0 + p[HID + tid] * e1 + p[2 * HID + tid] * e2 + p[3 * HID + tid] * e3) * inv;
}

// ---------------------------------------------------------------------------
extern "C" void kernel_launch(void* const* d_in, const int* in_sizes, int n_in,
                              void* d_out, int out_size, void* d_ws, size_t ws_size,
                              hipStream_t stream)
{
    const int N = NN, E = EE;
    const float* x        = (const float*)d_in[0];
    const float* edgeattr = (const float*)d_in[1];
    const float* time_emb = (const float*)d_in[2];
    const float* W_enc1   = (const float*)d_in[3];
    const float* b_enc1   = (const float*)d_in[4];
    const float* g_enc    = (const float*)d_in[5];
    const float* be_enc   = (const float*)d_in[6];
    const float* W_enc2   = (const float*)d_in[7];
    const float* b_enc2   = (const float*)d_in[8];
    const float* W_c[3]   = {(const float*)d_in[9],  (const float*)d_in[11], (const float*)d_in[13]};
    const float* b_c[3]   = {(const float*)d_in[10], (const float*)d_in[12], (const float*)d_in[14]};
    const float* ln_g     = (const float*)d_in[15];
    const float* ln_b     = (const float*)d_in[16];
    const float* W_a1     = (const float*)d_in[17];
    const float* b_a1     = (const float*)d_in[18];
    const float* W_a2     = (const float*)d_in[19];
    const float* b_a2     = (const float*)d_in[20];
    const float* W_g      = (const float*)d_in[21];
    const float* b_g      = (const float*)d_in[22];
    const int*   eidx     = (const int*)d_in[23];
    float* out            = (float*)d_out;

    // workspace layout
    char* ws = (char*)d_ws;
    size_t off = 0;
    auto alloc = [&](size_t bytes) -> char* {
        char* p = ws + off; off += (bytes + 255) & ~(size_t)255; return p;
    };
    __bf16* wt_enc1 = (__bf16*)alloc((size_t)256 * 64 * 2);
    __bf16* wt_enc2 = (__bf16*)alloc((size_t)224 * 256 * 2);
    __bf16* wt_c0   = (__bf16*)alloc((size_t)256 * 288 * 2);
    __bf16* wt_c1   = (__bf16*)alloc((size_t)256 * 288 * 2);
    __bf16* wt_c2   = (__bf16*)alloc((size_t)256 * 288 * 2);
    __bf16* wt_a1   = (__bf16*)alloc((size_t)128 * 256 * 2);
    __bf16* wt_cv[3] = {wt_c0, wt_c1, wt_c2};
    __bf16* x_bf    = (__bf16*)alloc((size_t)N * 64 * 2);
    float*  h1      = (float*)alloc((size_t)N * HID * 4);   // reused as conv pre-activation
    __bf16* h1n     = (__bf16*)alloc((size_t)N * HID * 2);  // reused as s1 (fp32 [N,128])
    float*  h       = (float*)alloc((size_t)N * HID * 4);
    __bf16* hb      = (__bf16*)alloc((size_t)N * HID * 2);
    float*  agg     = (float*)alloc((size_t)N * HID * 4);
    __bf16* aggb    = (__bf16*)alloc((size_t)N * 288 * 2);
    float*  eagg    = (float*)alloc((size_t)N * 32 * 4);
    float*  deg     = (float*)alloc((size_t)N * 4);
    float*  score   = (float*)alloc((size_t)N * 4);
    float*  pooled  = (float*)alloc((size_t)2048 * HID * 4);
    float*  pre     = h1;                  // alias (h1 dead after encoder LN)
    float*  s1      = (float*)h1n;         // alias (h1n dead after enc2 GEMM)

    // ---- invariant prep -------------------------------------------------
    hipMemsetAsync(deg,  0, (size_t)N * 4,       stream);
    hipMemsetAsync(eagg, 0, (size_t)N * 32 * 4,  stream);

    transcvt_k<<<(64 * 256 + 255) / 256, 256, 0, stream>>>(W_enc1, wt_enc1, 64, 256);
    transcvt_k<<<(256 * 224 + 255) / 256, 256, 0, stream>>>(W_enc2, wt_enc2, 256, 224);
    for (int i = 0; i < 3; ++i)
        transcvt_k<<<(288 * 256 + 255) / 256, 256, 0, stream>>>(W_c[i], wt_cv[i], 288, 256);
    transcvt_k<<<(256 * 128 + 255) / 256, 256, 0, stream>>>(W_a1, wt_a1, 256, 128);

    f2bf_k<<<(N * 64 + 255) / 256, 256, 0, stream>>>(x, x_bf, N * 64);

    edge_deg_k<<<E / 8, 256, 0, stream>>>(eidx, edgeattr, eagg, deg, E);
    eagg_cat_k<<<N * 32 / 256, 256, 0, stream>>>(eagg, aggb);

    // ---- encoder --------------------------------------------------------
    // h1 = relu(x @ W_enc1 + b)    [N,64]x[64,256]
    {
        int groups = (N / 16) * ((256 / 16 + 3) / 4);
        wmma_gemm_k<4><<<(groups + 7) / 8, 256, 0, stream>>>(
            x_bf, wt_enc1, b_enc1, nullptr, h1, N, 64, 256, HID, EPI_RELU);
    }
    ln_enc_k<<<N / 8, 256, 0, stream>>>(h1, g_enc, be_enc, h1n, N);
    // h[:,0:224] = h1n @ W_enc2 + b ; h[:,224:256] = time_emb[batch]
    {
        int groups = (N / 16) * ((224 / 16 + 3) / 4);
        wmma_gemm_k<4><<<(groups + 7) / 8, 256, 0, stream>>>(
            h1n, wt_enc2, b_enc2, nullptr, h, N, 256, 224, HID, EPI_NONE);
    }
    timecat_k<<<N * 32 / 256, 256, 0, stream>>>(time_emb, h);

    // ---- 3 conv layers --------------------------------------------------
    for (int layer = 0; layer < 3; ++layer) {
        hipMemsetAsync(agg, 0, (size_t)N * HID * 4, stream);
        scatter_k<<<E / 8, 256, 0, stream>>>(eidx, h, agg, E);
        aggcvt_k<<<N * HID / 256, 256, 0, stream>>>(agg, aggb);
        int groups = (N / 16) * ((256 / 16 + 3) / 4);
        wmma_gemm_k<4><<<(groups + 7) / 8, 256, 0, stream>>>(
            aggb, wt_cv[layer], b_c[layer], deg, pre, N, 288, 256, HID, EPI_DEG);
        ln_silu_res_k<<<N / 8, 256, 0, stream>>>(
            pre, ln_g + layer * HID, ln_b + layer * HID, h, hb, N, layer == 0 ? 1 : 0);
    }

    // ---- attention pooling ---------------------------------------------
    {
        int groups = (N / 16) * ((128 / 16 + 3) / 4);
        wmma_gemm_k<4><<<(groups + 7) / 8, 256, 0, stream>>>(
            hb, wt_a1, b_a1, nullptr, s1, N, 256, 128, 128, EPI_TANH);
    }
    score_k<<<N / 8, 256, 0, stream>>>(s1, W_a2, b_a2, score, N);
    pool_k<<<2048, 256, 0, stream>>>(h, score, pooled);
    fuse_k<<<512, 256, 0, stream>>>(pooled, W_g, b_g, out);

    (void)in_sizes; (void)n_in; (void)out_size; (void)ws_size;
}
